// TemporalBand_56762287784345
// MI455X (gfx1250) — matrix operand order repaired and checked by reading, not compiled, not run on hardware
//
#include <hip/hip_runtime.h>
#include <stdint.h>

typedef __attribute__((ext_vector_type(16))) __bf16 v16bf;
typedef __attribute__((ext_vector_type(8)))  float  v8f;
typedef unsigned int u32x4 __attribute__((ext_vector_type(4)));
typedef int          i32x8 __attribute__((ext_vector_type(8)));
typedef int          i32x4 __attribute__((ext_vector_type(4)));

union BF16x16 { v16bf v; uint4 q[2]; };

static __device__ __forceinline__ uint16_t f2bf(float f) {
    union { float f; unsigned u; } a; a.f = f;
    unsigned u = a.u;
    unsigned r = u + 0x7FFFu + ((u >> 16) & 1u);   // round-to-nearest-even
    return (uint16_t)(r >> 16);
}

static __device__ __forceinline__ v8f wmma_bf16(v16bf a, v16bf b, v8f c) {
    return __builtin_amdgcn_wmma_f32_16x16x32_bf16(false, a, false, b, (short)0, c, false, false);
}

#define LDP 40            // padded LDS row stride in bf16 elems (64B row + 16B pad)
#define TILE_BYTES (128 * LDP * 2)

// Issue one TDM 2-D tile load: 128 rows x 32 bf16, row stride `stride0` elems,
// LDS-padded to LDP via pad_interval=64B / pad_amount=16B.
static __device__ __forceinline__ void tdm_tile_load(uint32_t lds_off, const uint16_t* gptr,
                                                     uint32_t stride0) {
    uint64_t ga = (uint64_t)(uintptr_t)gptr;
    u32x4 g0;
    g0[0] = 1u;                                            // count=1, no gather
    g0[1] = lds_off;                                       // lds_addr
    g0[2] = (uint32_t)ga;                                  // global_addr[31:0]
    g0[3] = (uint32_t)((ga >> 32) & 0x1FFFFFFu) | (2u << 30); // addr[56:32] | type=2
    i32x8 g1;
    g1[0] = (int)((1u << 16)        // data_size = 2 bytes
                | (1u << 20)        // pad_enable
                | (3u << 22)        // pad_interval: 2^(3+1)=16 DW = 64B (one row)
                | (3u << 25));      // pad_amount: 3+1 = 4 DW = 16B
    g1[1] = (int)((stride0 & 0xFFFFu) << 16);              // tensor_dim0[15:0]
    g1[2] = (int)(((stride0 >> 16) & 0xFFFFu)              // tensor_dim0[31:16]
                | (0x2000u << 16));                        // tensor_dim1[15:0] = 8192
    g1[3] = (int)(32u << 16);                              // tensor_dim1 hi | tile_dim0=32
    g1[4] = (int)128u;                                     // tile_dim1=128, tile_dim2=0
    g1[5] = (int)stride0;                                  // tensor_dim0_stride[31:0]
    g1[6] = 0;                                             // stride hi | dim1_stride lo
    g1[7] = 0;
    i32x4 z4 = { 0, 0, 0, 0 };
    i32x8 z8 = { 0, 0, 0, 0, 0, 0, 0, 0 };
    __builtin_amdgcn_tensor_load_to_lds(g0, g1, z4, z4, z8, 0);
}

// ---------------------------------------------------------------- weight fp32 -> bf16
__global__ __launch_bounds__(256) void wconv(const float* __restrict__ src,
                                             uint16_t* __restrict__ dst) {
    const size_t i = ((size_t)blockIdx.x * 256 + threadIdx.x) * 8;
    float4 f0 = *(const float4*)(src + i);
    float4 f1 = *(const float4*)(src + i + 4);
    uint16_t h[8] = { f2bf(f0.x), f2bf(f0.y), f2bf(f0.z), f2bf(f0.w),
                      f2bf(f1.x), f2bf(f1.y), f2bf(f1.z), f2bf(f1.w) };
    *(uint4*)(dst + i) = *(const uint4*)h;
}

// ---------------------------------------------------------------- LayerNorm -> bf16
__global__ __launch_bounds__(128) void ln_kernel(
    const float* __restrict__ x, const float* __restrict__ gamma,
    const float* __restrict__ beta, uint16_t* __restrict__ Xn)
{
    const int row = blockIdx.x;
    const int tid = threadIdx.x;
    const float* xr = x + (size_t)row * 1024 + tid * 8;

    float4 v0 = *(const float4*)(xr);
    float4 v1 = *(const float4*)(xr + 4);
    float s  = v0.x + v0.y + v0.z + v0.w + v1.x + v1.y + v1.z + v1.w;
    float sq = v0.x*v0.x + v0.y*v0.y + v0.z*v0.z + v0.w*v0.w
             + v1.x*v1.x + v1.y*v1.y + v1.z*v1.z + v1.w*v1.w;
    #pragma unroll
    for (int m = 1; m < 32; m <<= 1) {
        s  += __shfl_xor(s,  m, 32);
        sq += __shfl_xor(sq, m, 32);
    }
    __shared__ float sh[8];
    const int wid = tid >> 5, lane = tid & 31;
    if (lane == 0) { sh[wid] = s; sh[4 + wid] = sq; }
    __syncthreads();
    s  = sh[0] + sh[1] + sh[2] + sh[3];
    sq = sh[4] + sh[5] + sh[6] + sh[7];
    const float mu  = s * (1.0f / 1024.0f);
    const float var = sq * (1.0f / 1024.0f) - mu * mu;
    const float inv = rsqrtf(var + 1e-5f);

    float4 g0 = *(const float4*)(gamma + tid * 8);
    float4 g1 = *(const float4*)(gamma + tid * 8 + 4);
    float4 b0 = *(const float4*)(beta  + tid * 8);
    float4 b1 = *(const float4*)(beta  + tid * 8 + 4);

    uint16_t h[8];
    h[0] = f2bf((v0.x - mu) * inv * g0.x + b0.x);
    h[1] = f2bf((v0.y - mu) * inv * g0.y + b0.y);
    h[2] = f2bf((v0.z - mu) * inv * g0.z + b0.z);
    h[3] = f2bf((v0.w - mu) * inv * g0.w + b0.w);
    h[4] = f2bf((v1.x - mu) * inv * g1.x + b1.x);
    h[5] = f2bf((v1.y - mu) * inv * g1.y + b1.y);
    h[6] = f2bf((v1.z - mu) * inv * g1.z + b1.z);
    h[7] = f2bf((v1.w - mu) * inv * g1.w + b1.w);
    *(uint4*)(Xn + (size_t)row * 1024 + tid * 8) = *(const uint4*)h;
}

// ---------------------------------------------------------------- fused QKV GEMM (TDM)
// C[8192,3072] = Xn[8192,1024] @ [Wq;Wk;Wv]^T + bias  -> bf16 Q/K/V
__global__ __launch_bounds__(256) void qkv_gemm(
    const uint16_t* __restrict__ Xn,
    const uint16_t* __restrict__ Wqb, const uint16_t* __restrict__ Wkb,
    const uint16_t* __restrict__ Wvb,
    const float* __restrict__ bq, const float* __restrict__ bk,
    const float* __restrict__ bv,
    uint16_t* __restrict__ Qb, uint16_t* __restrict__ Kb, uint16_t* __restrict__ Vb)
{
    __shared__ uint16_t As[2][128 * LDP];
    __shared__ uint16_t Bs[2][128 * LDP];
    const int K = 1024;
    const int bn = blockIdx.x;                     // 0..23 over N=3072
    const int bm = blockIdx.y;                     // 0..63 over M=8192
    const int tid = threadIdx.x;
    const int wid = tid >> 5, lane = tid & 31;
    const int l15 = lane & 15, lh = (lane >> 4) & 1;
    const int wM = (wid & 3) * 32;
    const int wN = (wid >> 2) * 64;
    const uint16_t* Wsel = (bn < 8) ? Wqb : (bn < 16) ? Wkb : Wvb;
    const int bnl = bn & 7;
    const uint16_t* Abase = Xn   + (size_t)(bm * 128)  * K;
    const uint16_t* Bbase = Wsel + (size_t)(bnl * 128) * K;
    const uint32_t asOff = (uint32_t)(uintptr_t)(void*)&As[0][0];
    const uint32_t bsOff = (uint32_t)(uintptr_t)(void*)&Bs[0][0];
    const bool issuer = (wid == 0);

    v8f acc[2][4];
    #pragma unroll
    for (int i = 0; i < 2; ++i)
        #pragma unroll
        for (int j = 0; j < 4; ++j) acc[i][j] = (v8f)0.0f;

    if (issuer) {                                   // prime buffer 0
        tdm_tile_load(asOff, Abase, 1024);
        tdm_tile_load(bsOff, Bbase, 1024);
    }
    for (int k0 = 0; k0 < K; k0 += 32) {
        const int buf = (k0 >> 5) & 1;
        if (issuer) {
            if (k0 + 32 < K) {                      // stream next tile into buf^1
                tdm_tile_load(asOff + (buf ^ 1) * TILE_BYTES, Abase + k0 + 32, 1024);
                tdm_tile_load(bsOff + (buf ^ 1) * TILE_BYTES, Bbase + k0 + 32, 1024);
                __builtin_amdgcn_s_wait_tensorcnt(2);   // current buf done, next in flight
            } else {
                __builtin_amdgcn_s_wait_tensorcnt(0);
            }
        }
        __syncthreads();

        const uint16_t* Ap = &As[buf][0];
        const uint16_t* Bp = &Bs[buf][0];
        BF16x16 a[2], b[4];
        #pragma unroll
        for (int i = 0; i < 2; ++i) {               // A frag 16x32
            const uint16_t* p = Ap + (wM + i * 16 + l15) * LDP + lh * 8;
            a[i].q[0] = *(const uint4*)p;
            a[i].q[1] = *(const uint4*)(p + 16);
        }
        #pragma unroll
        for (int j = 0; j < 4; ++j) {               // B frag 32x16 (K-contig)
            const uint16_t* p = Bp + (wN + j * 16 + l15) * LDP + lh * 8;
            b[j].q[0] = *(const uint4*)p;
            b[j].q[1] = *(const uint4*)(p + 16);
        }
        #pragma unroll
        for (int i = 0; i < 2; ++i)
            #pragma unroll
            for (int j = 0; j < 4; ++j)
                acc[i][j] = wmma_bf16(a[i].v, b[j].v, acc[i][j]);
        __syncthreads();                            // all reads of buf done before reuse
    }

    const float* bias = (bn < 8) ? bq : (bn < 16) ? bk : bv;
    uint16_t* Out = (bn < 8) ? Qb : (bn < 16) ? Kb : Vb;
    #pragma unroll
    for (int i = 0; i < 2; ++i)
        #pragma unroll
        for (int j = 0; j < 4; ++j) {
            const int n = bnl * 128 + wN + j * 16 + l15;
            const float bb = bias[n];
            #pragma unroll
            for (int e = 0; e < 8; ++e) {
                const int m = bm * 128 + wM + i * 16 + e + lh * 8;
                Out[(size_t)m * 1024 + n] = f2bf(acc[i][j][e] + bb);
            }
        }
}

// ---------------------------------------------------------------- flash attention
__global__ __launch_bounds__(128) void attn_kernel(
    const uint16_t* __restrict__ Qb, const uint16_t* __restrict__ Kb,
    const uint16_t* __restrict__ Vb, uint16_t* __restrict__ Ob)
{
    const int T = 2048, D = 1024, HD = 64;
    __shared__ uint16_t Ps[64 * 72];
    __shared__ uint16_t Vt[64 * 72];   // V transposed: [d][n]
    const int qblk = blockIdx.x, h = blockIdx.y, b = blockIdx.z;
    const int tid = threadIdx.x, wid = tid >> 5, lane = tid & 31;
    const int l15 = lane & 15, lh = (lane >> 4) & 1;
    const size_t rowbase = (size_t)b * T;
    const int hoff = h * HD;

    BF16x16 qf[2];
    const int qrow = qblk * 64 + wid * 16 + l15;
    #pragma unroll
    for (int d = 0; d < 2; ++d) {
        const uint16_t* p = Qb + (rowbase + qrow) * D + hoff + d * 32 + lh * 8;
        qf[d].q[0] = *(const uint4*)p;
        qf[d].q[1] = *(const uint4*)(p + 16);
    }

    v8f o[4];
    #pragma unroll
    for (int t = 0; t < 4; ++t) o[t] = (v8f)0.0f;
    float mrow[8], lrow[8];
    #pragma unroll
    for (int e = 0; e < 8; ++e) { mrow[e] = -1e30f; lrow[e] = 0.0f; }

    const int nblocks = qblk + 1;                   // causal
    for (int kb = 0; kb < nblocks; ++kb) {
        const int n0 = kb * 64;
        if (kb + 1 < nblocks) {                     // prefetch next KV block rows
            const size_t pr = (rowbase + n0 + 64 + lane) * D + hoff;
            __builtin_prefetch(Kb + pr, 0, 0);
            __builtin_prefetch(Kb + pr + 32 * D, 0, 0);
            __builtin_prefetch(Vb + pr, 0, 0);
            __builtin_prefetch(Vb + pr + 32 * D, 0, 0);
        }
        __syncthreads();                            // protect Ps/Vt reuse
        #pragma unroll
        for (int it = 0; it < 4; ++it) {            // stage V transposed
            int t = tid + it * 128;
            int n = t >> 3, dg = (t & 7) * 8;
            uint4 vv = *(const uint4*)(Vb + (rowbase + n0 + n) * D + hoff + dg);
            const uint16_t* hp = (const uint16_t*)&vv;
            #pragma unroll
            for (int i = 0; i < 8; ++i) Vt[(dg + i) * 72 + n] = hp[i];
        }

        v8f s[4];
        #pragma unroll
        for (int j = 0; j < 4; ++j) s[j] = (v8f)0.0f;
        #pragma unroll
        for (int d = 0; d < 2; ++d)                 // S = Q @ K^T
            #pragma unroll
            for (int j = 0; j < 4; ++j) {
                const int n = n0 + j * 16 + l15;
                const uint16_t* p = Kb + (rowbase + n) * D + hoff + d * 32 + lh * 8;
                BF16x16 kf;
                kf.q[0] = *(const uint4*)p;
                kf.q[1] = *(const uint4*)(p + 16);
                s[j] = wmma_bf16(qf[d].v, kf.v, s[j]);
            }

        const int rg = qblk * 64 + wid * 16 + lh * 8;   // + e = global q row
        #pragma unroll
        for (int j = 0; j < 4; ++j) {
            const int cg = n0 + j * 16 + l15;
            #pragma unroll
            for (int e = 0; e < 8; ++e) {
                float v = s[j][e] * 0.125f;             // 1/sqrt(64)
                s[j][e] = (cg > rg + e) ? -1e9f : v;
            }
        }
        #pragma unroll
        for (int e = 0; e < 8; ++e) {                   // online softmax
            float mx = fmaxf(fmaxf(s[0][e], s[1][e]), fmaxf(s[2][e], s[3][e]));
            #pragma unroll
            for (int m = 1; m < 16; m <<= 1) mx = fmaxf(mx, __shfl_xor(mx, m, 32));
            const float mnew = fmaxf(mrow[e], mx);
            const float sc = __expf(mrow[e] - mnew);
            float rs = 0.0f;
            #pragma unroll
            for (int j = 0; j < 4; ++j) {
                float p = __expf(s[j][e] - mnew);
                s[j][e] = p; rs += p;
            }
            #pragma unroll
            for (int m = 1; m < 16; m <<= 1) rs += __shfl_xor(rs, m, 32);
            lrow[e] = lrow[e] * sc + rs;
            mrow[e] = mnew;
            #pragma unroll
            for (int t = 0; t < 4; ++t) o[t][e] *= sc;
        }
        #pragma unroll
        for (int j = 0; j < 4; ++j)                     // P -> LDS (bf16)
            #pragma unroll
            for (int e = 0; e < 8; ++e)
                Ps[(wid * 16 + e + lh * 8) * 72 + j * 16 + l15] = f2bf(s[j][e]);
        __syncthreads();

        #pragma unroll
        for (int ks = 0; ks < 2; ++ks) {                // O += P @ V
            const uint16_t* ap = Ps + (wid * 16 + l15) * 72 + ks * 32 + lh * 8;
            BF16x16 af;
            af.q[0] = *(const uint4*)ap;
            af.q[1] = *(const uint4*)(ap + 16);
            #pragma unroll
            for (int ct = 0; ct < 4; ++ct) {
                const uint16_t* bp = Vt + (ct * 16 + l15) * 72 + ks * 32 + lh * 8;
                BF16x16 vf;
                vf.q[0] = *(const uint4*)bp;
                vf.q[1] = *(const uint4*)(bp + 16);
                o[ct] = wmma_bf16(af.v, vf.v, o[ct]);
            }
        }
    }

    #pragma unroll
    for (int ct = 0; ct < 4; ++ct)
        #pragma unroll
        for (int e = 0; e < 8; ++e) {
            const float inv = 1.0f / lrow[e];
            const size_t m = rowbase + qblk * 64 + wid * 16 + e + lh * 8;
            Ob[m * D + hoff + ct * 16 + l15] = f2bf(o[ct][e] * inv);
        }
}

// ---------------------------------------------------------------- out projection + residual (TDM)
__global__ __launch_bounds__(256) void out_gemm(
    const uint16_t* __restrict__ Ob, const uint16_t* __restrict__ Wob,
    const float* __restrict__ bo, const float* __restrict__ x,
    float* __restrict__ out)
{
    __shared__ uint16_t As[2][128 * LDP];
    __shared__ uint16_t Bs[2][128 * LDP];
    const int K = 1024;
    const int bn = blockIdx.x;                     // 0..7
    const int bm = blockIdx.y;                     // 0..63
    const int tid = threadIdx.x;
    const int wid = tid >> 5, lane = tid & 31;
    const int l15 = lane & 15, lh = (lane >> 4) & 1;
    const int wM = (wid & 3) * 32;
    const int wN = (wid >> 2) * 64;
    const uint16_t* Abase = Ob  + (size_t)(bm * 128) * K;
    const uint16_t* Bbase = Wob + (size_t)(bn * 128) * K;
    const uint32_t asOff = (uint32_t)(uintptr_t)(void*)&As[0][0];
    const uint32_t bsOff = (uint32_t)(uintptr_t)(void*)&Bs[0][0];
    const bool issuer = (wid == 0);

    v8f acc[2][4];
    #pragma unroll
    for (int i = 0; i < 2; ++i)
        #pragma unroll
        for (int j = 0; j < 4; ++j) acc[i][j] = (v8f)0.0f;

    if (issuer) {
        tdm_tile_load(asOff, Abase, 1024);
        tdm_tile_load(bsOff, Bbase, 1024);
    }
    for (int k0 = 0; k0 < K; k0 += 32) {
        const int buf = (k0 >> 5) & 1;
        if (issuer) {
            if (k0 + 32 < K) {
                tdm_tile_load(asOff + (buf ^ 1) * TILE_BYTES, Abase + k0 + 32, 1024);
                tdm_tile_load(bsOff + (buf ^ 1) * TILE_BYTES, Bbase + k0 + 32, 1024);
                __builtin_amdgcn_s_wait_tensorcnt(2);
            } else {
                __builtin_amdgcn_s_wait_tensorcnt(0);
            }
        }
        __syncthreads();

        const uint16_t* Ap = &As[buf][0];
        const uint16_t* Bp = &Bs[buf][0];
        BF16x16 a[2], b[4];
        #pragma unroll
        for (int i = 0; i < 2; ++i) {
            const uint16_t* p = Ap + (wM + i * 16 + l15) * LDP + lh * 8;
            a[i].q[0] = *(const uint4*)p;
            a[i].q[1] = *(const uint4*)(p + 16);
        }
        #pragma unroll
        for (int j = 0; j < 4; ++j) {
            const uint16_t* p = Bp + (wN + j * 16 + l15) * LDP + lh * 8;
            b[j].q[0] = *(const uint4*)p;
            b[j].q[1] = *(const uint4*)(p + 16);
        }
        #pragma unroll
        for (int i = 0; i < 2; ++i)
            #pragma unroll
            for (int j = 0; j < 4; ++j)
                acc[i][j] = wmma_bf16(a[i].v, b[j].v, acc[i][j]);
        __syncthreads();
    }

    #pragma unroll
    for (int i = 0; i < 2; ++i)
        #pragma unroll
        for (int j = 0; j < 4; ++j) {
            const int n = bn * 128 + wN + j * 16 + l15;
            const float bb = bo[n];
            #pragma unroll
            for (int e = 0; e < 8; ++e) {
                const size_t m = (size_t)bm * 128 + wM + i * 16 + e + lh * 8;
                out[m * 1024 + n] = acc[i][j][e] + bb + x[m * 1024 + n];
            }
        }
}

// ---------------------------------------------------------------- launch
extern "C" void kernel_launch(void* const* d_in, const int* in_sizes, int n_in,
                              void* d_out, int out_size, void* d_ws, size_t ws_size,
                              hipStream_t stream) {
    const float* x     = (const float*)d_in[0];
    const float* Wq    = (const float*)d_in[1];
    const float* bq    = (const float*)d_in[2];
    const float* Wk    = (const float*)d_in[3];
    const float* bk    = (const float*)d_in[4];
    const float* Wv    = (const float*)d_in[5];
    const float* bv    = (const float*)d_in[6];
    const float* Wo    = (const float*)d_in[7];
    const float* bo    = (const float*)d_in[8];
    const float* gamma = (const float*)d_in[9];
    const float* beta  = (const float*)d_in[10];
    float* out = (float*)d_out;

    uint8_t* ws = (uint8_t*)d_ws;
    uint16_t* Xn  = (uint16_t*)(ws);                        // 16 MB
    uint16_t* Qb  = (uint16_t*)(ws + ((size_t)16 << 20));   // 16 MB
    uint16_t* Kb  = (uint16_t*)(ws + ((size_t)32 << 20));   // 16 MB
    uint16_t* Vb  = (uint16_t*)(ws + ((size_t)48 << 20));   // 16 MB
    uint16_t* Ob  = (uint16_t*)(ws + ((size_t)64 << 20));   // 16 MB
    uint16_t* Wqb = (uint16_t*)(ws + ((size_t)80 << 20));   // 2 MB each
    uint16_t* Wkb = (uint16_t*)(ws + ((size_t)82 << 20));
    uint16_t* Wvb = (uint16_t*)(ws + ((size_t)84 << 20));
    uint16_t* Wob = (uint16_t*)(ws + ((size_t)86 << 20));

    wconv<<<512, 256, 0, stream>>>(Wq, Wqb);
    wconv<<<512, 256, 0, stream>>>(Wk, Wkb);
    wconv<<<512, 256, 0, stream>>>(Wv, Wvb);
    wconv<<<512, 256, 0, stream>>>(Wo, Wob);
    ln_kernel  <<<8192, 128, 0, stream>>>(x, gamma, beta, Xn);
    qkv_gemm   <<<dim3(24, 64), 256, 0, stream>>>(Xn, Wqb, Wkb, Wvb, bq, bk, bv, Qb, Kb, Vb);
    attn_kernel<<<dim3(32, 16, 4), 128, 0, stream>>>(Qb, Kb, Vb, Ob);
    out_gemm   <<<dim3(8, 64), 256, 0, stream>>>(Ob, Wob, bo, x, out);
}
